// KVCached_MHSA_90615220011030
// MI455X (gfx1250) — matrix operand order repaired and checked by reading, not compiled, not run on hardware
//
#include <hip/hip_runtime.h>
#include <hip/hip_bf16.h>

#define BQ   2
#define SEQ  2048
#define EMB  2048
#define NH   16
#define HD   128

typedef __attribute__((ext_vector_type(16))) __bf16 v16bf;
typedef __attribute__((ext_vector_type(8)))  __bf16 v8bf;
typedef __attribute__((ext_vector_type(8)))  float  v8f;
typedef uint32_t u32x4 __attribute__((ext_vector_type(4)));
typedef uint32_t u32x8 __attribute__((ext_vector_type(8)));

static __device__ __forceinline__ v8f wmma_bf16(v16bf a, v16bf b, v8f c) {
  // D = A(16x32 bf16) * B(32x16 bf16) + C(16x16 f32)
  return __builtin_amdgcn_wmma_f32_16x16x32_bf16(false, a, false, b, (short)0, c, false, false);
}

static __device__ __forceinline__ v16bf pack16(v8bf lo, v8bf hi) {
  v16bf r;
  #pragma unroll
  for (int i = 0; i < 8; ++i) { r[i] = lo[i]; r[8 + i] = hi[i]; }
  return r;
}

// ---------------------------------------------------------------------------
// TDM: 2D tile Global -> LDS, data_size = 2B, LDS padding 4 dwords / 64 dwords
// (effective LDS row stride = tile0 + 8 halfs). Groups 0+1 only (2D tensor).
// ---------------------------------------------------------------------------
static __device__ __forceinline__ void tdm_load_2d(uint32_t lds_byte_off,
                                                   const void* gptr,
                                                   uint32_t tdim0, uint32_t tdim1,
                                                   uint32_t tile0, uint32_t tile1,
                                                   uint64_t stride0) {
  const uint64_t ga = (uint64_t)gptr;
  u32x4 g0;
  g0[0] = 1u;                                            // count=1 (valid user D#)
  g0[1] = lds_byte_off;                                  // lds_addr
  g0[2] = (uint32_t)ga;                                  // global_addr[31:0]
  g0[3] = ((uint32_t)(ga >> 32) & 0x01FFFFFFu) | (2u << 30);  // addr[56:32], type=2
  u32x8 g1;
  g1[0] = (1u << 16)        // data_size = 1 (2 bytes)
        | (1u << 20)        // pad_enable
        | (5u << 22)        // pad_interval: 64 dwords
        | (3u << 25);       // pad_amount:   4 dwords (16B -> row stride +8 halfs)
  g1[1] = (tdim0 & 0xFFFFu) << 16;                            // tensor_dim0[15:0]
  g1[2] = ((tdim0 >> 16) & 0xFFFFu) | ((tdim1 & 0xFFFFu) << 16);
  g1[3] = ((tdim1 >> 16) & 0xFFFFu) | ((tile0 & 0xFFFFu) << 16);  // tile_dim0
  g1[4] = (tile1 & 0xFFFFu);                                  // tile_dim1, tile_dim2=0
  g1[5] = (uint32_t)stride0;                                  // tensor_dim0_stride lo
  g1[6] = (uint32_t)(stride0 >> 32) & 0xFFFFu;                // stride hi / dim1_stride=0
  g1[7] = 0u;
  asm volatile("tensor_load_to_lds %0, %1" :: "s"(g0), "s"(g1) : "memory");
}

// ---------------------------------------------------------------------------
// Kernel 1: QKV projection + bias + scale + RoPE.
//   grid = (B*SEQ/128, EMB/128, 3)   block = 256 (8 waves)
//   Q,K stored bf16 [B,H,L,HD]; V stored transposed bf16 [B,H,HD,L]
// ---------------------------------------------------------------------------
__launch_bounds__(256)
__global__ void qkv_rope_wmma(const float* __restrict__ x,
                              const float* __restrict__ Wq, const float* __restrict__ bq,
                              const float* __restrict__ Wk, const float* __restrict__ bk,
                              const float* __restrict__ Wv, const float* __restrict__ bv,
                              __bf16* __restrict__ qo, __bf16* __restrict__ ko,
                              __bf16* __restrict__ vTo)
{
  const int mat = blockIdx.z;                       // 0=Q 1=K 2=V
  const float* W    = (mat == 0) ? Wq : (mat == 1) ? Wk : Wv;
  const float* bias = (mat == 0) ? bq : (mat == 1) ? bk : bv;
  const float scale = (mat == 0) ? 0.08838834764831845f : 1.0f;  // HD^-0.5 on Q

  __shared__ union {
    struct { __bf16 A[128][40]; __bf16 BT[128][40]; } ab;  // 20.5 KB (BT transposed!)
    __bf16 C[128][136];                                    // epilogue staging 34.8 KB
  } sm;

  const int tid  = threadIdx.x;
  const int lane = tid & 31;
  const int wave = tid >> 5;
  const bool hiL = lane >= 16;
  const int ln   = lane & 15;

  const int row0 = blockIdx.x * 128;                // over B*SEQ
  const int col0 = blockIdx.y * 128;                // over EMB (== head*HD)
  const int wr = wave >> 1, wc = wave & 1;          // 4x2 wave grid -> 32x64 per wave

  v8f acc[2][4] = {};

  for (int k0 = 0; k0 < EMB; k0 += 32) {
    if (k0 + 32 < EMB) {
      __builtin_prefetch(x + (size_t)(row0 + (tid >> 1)) * EMB + k0 + 32, 0, 1);
      __builtin_prefetch(W + (size_t)(k0 + 32 + (tid >> 3)) * EMB + col0, 0, 1);
    }
    // --- A tile: x[row0..+128][k0..+32] f32 -> bf16 LDS, row-major
    #pragma unroll
    for (int it = 0; it < 4; ++it) {
      const int idx = tid * 4 + it;                 // 0..1023
      const int r = idx >> 3, c4 = idx & 7;         // 8 float4 per row
      const float4 v = *(const float4*)(x + (size_t)(row0 + r) * EMB + k0 + c4 * 4);
      sm.ab.A[r][c4 * 4 + 0] = (__bf16)v.x;  sm.ab.A[r][c4 * 4 + 1] = (__bf16)v.y;
      sm.ab.A[r][c4 * 4 + 2] = (__bf16)v.z;  sm.ab.A[r][c4 * 4 + 3] = (__bf16)v.w;
    }
    // --- B tile: W[k0..+32][col0..+128] f32 -> bf16 LDS, TRANSPOSED (BT[col][k])
    //     so B-fragments become two contiguous 8-half reads (2x ds_load_b128)
    #pragma unroll
    for (int it = 0; it < 4; ++it) {
      const int idx = tid * 4 + it;
      const int r = idx >> 5, c4 = idx & 31;        // 32 float4 per row
      const float4 v = *(const float4*)(W + (size_t)(k0 + r) * EMB + col0 + c4 * 4);
      sm.ab.BT[c4 * 4 + 0][r] = (__bf16)v.x;  sm.ab.BT[c4 * 4 + 1][r] = (__bf16)v.y;
      sm.ab.BT[c4 * 4 + 2][r] = (__bf16)v.z;  sm.ab.BT[c4 * 4 + 3][r] = (__bf16)v.w;
    }
    __syncthreads();

    // --- fragments + 8 WMMAs
    const int kA = hiL ? 8 : 0;                     // A halves own K {0..7,16..23}/{8..15,24..31}
    v16bf afrag[2];
    #pragma unroll
    for (int ti = 0; ti < 2; ++ti) {
      const __bf16* ap = &sm.ab.A[wr * 32 + ti * 16 + ln][kA];
      afrag[ti] = pack16(*(const v8bf*)ap, *(const v8bf*)(ap + 16));
    }
    const int kB = hiL ? 8 : 0;                     // B halves own K {0..7,16..23}/{8..15,24..31}
    #pragma unroll
    for (int tj = 0; tj < 4; ++tj) {
      const __bf16* bp = &sm.ab.BT[wc * 64 + tj * 16 + ln][kB];
      const v16bf bfrag = pack16(*(const v8bf*)bp, *(const v8bf*)(bp + 16));
      #pragma unroll
      for (int ti = 0; ti < 2; ++ti) acc[ti][tj] = wmma_bf16(afrag[ti], bfrag, acc[ti][tj]);
    }
    __syncthreads();
  }

  // --- stage (acc + bias) * scale into LDS for cross-lane RoPE pairing
  #pragma unroll
  for (int ti = 0; ti < 2; ++ti)
    #pragma unroll
    for (int tj = 0; tj < 4; ++tj)
      #pragma unroll
      for (int r = 0; r < 8; ++r) {
        const int rr = wr * 32 + ti * 16 + r + (hiL ? 8 : 0);
        const int cc = wc * 64 + tj * 16 + ln;
        sm.C[rr][cc] = (__bf16)((acc[ti][tj][r] + bias[col0 + cc]) * scale);
      }
  __syncthreads();

  // --- epilogue: RoPE (Q,K) / passthrough (V), scatter to attention layouts
  const int head = blockIdx.y;
  #pragma unroll 4
  for (int i = 0; i < 64; ++i) {
    const int e = tid + i * 256;                    // 0..16383
    const int r = e >> 7, d = e & 127;
    const int g = row0 + r;
    const int b = g >> 11;                          // / SEQ
    const int l = g & (SEQ - 1);
    if (mat < 2) {
      const int j = d & 63;
      const float f = (float)l * __powf(10000.0f, -(float)j * (1.0f / 64.0f));
      const float cs = __cosf(f), sn = __sinf(f);
      const float t1 = (float)sm.C[r][j], t2 = (float)sm.C[r][j + 64];
      const float val = (d < 64) ? (t1 * cs - t2 * sn) : (t1 * sn + t2 * cs);
      __bf16* dst = (mat == 0) ? qo : ko;
      dst[(((size_t)b * NH + head) * SEQ + l) * HD + d] = (__bf16)val;
    } else {
      vTo[(((size_t)b * NH + head) * HD + d) * SEQ + l] = sm.C[r][d];
    }
  }
}

// ---------------------------------------------------------------------------
// Kernel 2: causal flash attention with TDM-fed K/V tiles.
//   grid = (SEQ/128, B*NH)   block = 256 (8 waves, 16 query rows per wave)
//   dynamic LDS: Klds[128][136] | Vlds[128][136] | Plds[8][16][136] (104448 B)
//   (136-half row stride produced by TDM pad: 4 dwords per 64 dwords)
// ---------------------------------------------------------------------------
#define KV_STRIDE 136                               // halfs per LDS row (128 + pad 8)
#define KV_TILE_BYTES (128 * KV_STRIDE * 2)         // 34816

__launch_bounds__(256)
__global__ void flash_attn_wmma(const __bf16* __restrict__ q,
                                const __bf16* __restrict__ k,
                                const __bf16* __restrict__ vT,
                                __bf16* __restrict__ attn)
{
  extern __shared__ __bf16 smem2[];
  __bf16* Klds = smem2;                             // [128][136]
  __bf16* Vlds = smem2 + 128 * KV_STRIDE;           // [128][136]
  __bf16* Plds = smem2 + 2 * 128 * KV_STRIDE;       // [8][16][136]
  const uint32_t lds_base = (uint32_t)(uintptr_t)smem2;

  const int bh = blockIdx.y;
  const int i0 = blockIdx.x * 128;
  const int tid = threadIdx.x;
  const int lane = tid & 31, wave = tid >> 5;
  const bool hiL = lane >= 16;
  const int ln = lane & 15;

  const __bf16* qb = q  + (size_t)bh * SEQ * HD;
  const __bf16* kb = k  + (size_t)bh * SEQ * HD;
  const __bf16* vb = vT + (size_t)bh * HD * SEQ;

  // Q A-fragments resident: wave rows i0 + wave*16 .. +15, K = HD = 4x32
  v16bf aq[4];
  {
    const int row = i0 + wave * 16 + ln;
    const int kA = hiL ? 8 : 0;
    #pragma unroll
    for (int kt = 0; kt < 4; ++kt) {
      const __bf16* p = qb + (size_t)row * HD + kt * 32 + kA;
      aq[kt] = pack16(*(const v8bf*)p, *(const v8bf*)(p + 16));
    }
  }

  v8f O[8] = {};
  float mrow[8], lrow[8];
  #pragma unroll
  for (int r = 0; r < 8; ++r) { mrow[r] = -1e30f; lrow[r] = 0.0f; }

  for (int j0 = 0; j0 <= i0; j0 += 128) {
    __syncthreads();                                // K/V LDS free for reuse
    if (wave == 0) {
      // K tile: rows j0..j0+127 of K[bh], each HD=128 halfs, contiguous rows
      tdm_load_2d(lds_base, kb + (size_t)j0 * HD,
                  /*tdim0=*/HD, /*tdim1=*/SEQ - j0, /*tile0=*/128, /*tile1=*/128,
                  /*stride0=*/HD);
      // V tile: 128 rows (dims) x 128 keys, row stride SEQ (strided 2D tile)
      tdm_load_2d(lds_base + KV_TILE_BYTES, vb + j0,
                  /*tdim0=*/SEQ, /*tdim1=*/HD, /*tile0=*/128, /*tile1=*/128,
                  /*stride0=*/SEQ);
      __builtin_amdgcn_s_wait_tensorcnt(1);         // K landed (V may still fly)
    }
    __syncthreads();                                // publish K tile

    // --- scores S = Q * K^T for 16 rows x 128 key cols (32 WMMAs, LDS-fed)
    v8f sc[8];
    #pragma unroll
    for (int nt = 0; nt < 8; ++nt) {
      v8f s = {};
      #pragma unroll
      for (int kt = 0; kt < 4; ++kt) {
        const __bf16* p = Klds + (nt * 16 + ln) * KV_STRIDE + kt * 32 + (hiL ? 8 : 0);
        const v16bf bk = pack16(*(const v8bf*)p, *(const v8bf*)(p + 16));
        s = wmma_bf16(aq[kt], bk, s);
      }
      sc[nt] = s;
    }
    if (wave == 0) __builtin_amdgcn_s_wait_tensorcnt(0);  // V landed
    __syncthreads();                                // publish V tile

    const bool diag = (j0 == i0);
    // --- online softmax per owned row (C-layout: VGPR r <-> row, 16-lane half)
    #pragma unroll
    for (int r = 0; r < 8; ++r) {
      const int rowg = i0 + wave * 16 + r + (hiL ? 8 : 0);
      if (diag) {
        #pragma unroll
        for (int nt = 0; nt < 8; ++nt)
          if (j0 + nt * 16 + ln > rowg) sc[nt][r] = -1e30f;
      }
      float mb = sc[0][r];
      #pragma unroll
      for (int nt = 1; nt < 8; ++nt) mb = fmaxf(mb, sc[nt][r]);
      mb = fmaxf(mb, __shfl_xor(mb, 1));
      mb = fmaxf(mb, __shfl_xor(mb, 2));
      mb = fmaxf(mb, __shfl_xor(mb, 4));
      mb = fmaxf(mb, __shfl_xor(mb, 8));            // stays inside 16-lane half
      const float mnew  = fmaxf(mrow[r], mb);
      const float alpha = __expf(mrow[r] - mnew);
      float rs = 0.0f;
      #pragma unroll
      for (int nt = 0; nt < 8; ++nt) {
        const float p = __expf(sc[nt][r] - mnew);
        sc[nt][r] = p;
        rs += p;
      }
      rs += __shfl_xor(rs, 1); rs += __shfl_xor(rs, 2);
      rs += __shfl_xor(rs, 4); rs += __shfl_xor(rs, 8);
      lrow[r] = lrow[r] * alpha + rs;
      mrow[r] = mnew;
      #pragma unroll
      for (int ot = 0; ot < 8; ++ot) O[ot][r] *= alpha;
    }

    // --- C-layout P -> LDS (row-major) so it can be re-read in A-layout
    #pragma unroll
    for (int nt = 0; nt < 8; ++nt)
      #pragma unroll
      for (int r = 0; r < 8; ++r)
        Plds[(wave * 16 + r + (hiL ? 8 : 0)) * KV_STRIDE + nt * 16 + ln] =
            (__bf16)sc[nt][r];
    // same-wave LDS ops are in order; compiler inserts s_wait_dscnt

    // --- O += P * V  (V tile in LDS, contiguous B-fragments; 32 WMMAs)
    #pragma unroll
    for (int kt = 0; kt < 4; ++kt) {
      const __bf16* pp = Plds + (wave * 16 + ln) * KV_STRIDE + kt * 32 + (hiL ? 8 : 0);
      const v16bf ap = pack16(*(const v8bf*)pp, *(const v8bf*)(pp + 16));
      #pragma unroll
      for (int ot = 0; ot < 8; ++ot) {
        const __bf16* vp = Vlds + (ot * 16 + ln) * KV_STRIDE + kt * 32 + (hiL ? 8 : 0);
        const v16bf bv = pack16(*(const v8bf*)vp, *(const v8bf*)(vp + 16));
        O[ot] = wmma_bf16(ap, bv, O[ot]);
      }
    }
  }

  // --- normalize and write attn output bf16 [B][SEQ][EMB]
  const int b = bh / NH, h = bh % NH;
  #pragma unroll
  for (int r = 0; r < 8; ++r) {
    const int rowg = i0 + wave * 16 + r + (hiL ? 8 : 0);
    const float inv = 1.0f / lrow[r];
    #pragma unroll
    for (int ot = 0; ot < 8; ++ot)
      attn[((size_t)b * SEQ + rowg) * EMB + h * HD + ot * 16 + ln] =
          (__bf16)(O[ot][r] * inv);
  }
}

// ---------------------------------------------------------------------------
// Kernel 3: output projection  out = attn(bf16) @ Wo(f32->bf16) + bo  (f32 out)
//   grid = (B*SEQ/128, EMB/128)   block = 256
// ---------------------------------------------------------------------------
__launch_bounds__(256)
__global__ void out_proj_wmma(const __bf16* __restrict__ a,
                              const float* __restrict__ Wo,
                              const float* __restrict__ bo,
                              float* __restrict__ out)
{
  __shared__ __bf16 As[128][40];
  __shared__ __bf16 BsT[128][40];                   // transposed: BT[col][k]

  const int tid = threadIdx.x;
  const int lane = tid & 31, wave = tid >> 5;
  const bool hiL = lane >= 16;
  const int ln = lane & 15;
  const int row0 = blockIdx.x * 128, col0 = blockIdx.y * 128;
  const int wr = wave >> 1, wc = wave & 1;

  v8f acc[2][4] = {};

  for (int k0 = 0; k0 < EMB; k0 += 32) {
    if (k0 + 32 < EMB) {
      __builtin_prefetch(a + (size_t)(row0 + (tid >> 1)) * EMB + k0 + 32, 0, 1);
      __builtin_prefetch(Wo + (size_t)(k0 + 32 + (tid >> 3)) * EMB + col0, 0, 1);
    }
    // A tile already bf16: 512 x 16B chunks, 2/thread
    #pragma unroll
    for (int it = 0; it < 2; ++it) {
      const int idx = tid * 2 + it;                 // 0..511
      const int r = idx >> 2, c = (idx & 3) * 8;    // 4 chunks per 32-wide row
      *(v8bf*)&As[r][c] = *(const v8bf*)(a + (size_t)(row0 + r) * EMB + k0 + c);
    }
    // B tile f32 -> bf16, transposed
    #pragma unroll
    for (int it = 0; it < 4; ++it) {
      const int idx = tid * 4 + it;
      const int r = idx >> 5, c4 = idx & 31;
      const float4 v = *(const float4*)(Wo + (size_t)(k0 + r) * EMB + col0 + c4 * 4);
      BsT[c4 * 4 + 0][r] = (__bf16)v.x;  BsT[c4 * 4 + 1][r] = (__bf16)v.y;
      BsT[c4 * 4 + 2][r] = (__bf16)v.z;  BsT[c4 * 4 + 3][r] = (__bf16)v.w;
    }
    __syncthreads();

    const int kA = hiL ? 8 : 0;
    v16bf afrag[2];
    #pragma unroll
    for (int ti = 0; ti < 2; ++ti) {
      const __bf16* ap = &As[wr * 32 + ti * 16 + ln][kA];
      afrag[ti] = pack16(*(const v8bf*)ap, *(const v8bf*)(ap + 16));
    }
    const int kB = hiL ? 8 : 0;
    #pragma unroll
    for (int tj = 0; tj < 4; ++tj) {
      const __bf16* bp = &BsT[wc * 64 + tj * 16 + ln][kB];
      const v16bf bfrag = pack16(*(const v8bf*)bp, *(const v8bf*)(bp + 16));
      #pragma unroll
      for (int ti = 0; ti < 2; ++ti) acc[ti][tj] = wmma_bf16(afrag[ti], bfrag, acc[ti][tj]);
    }
    __syncthreads();
  }

  #pragma unroll
  for (int ti = 0; ti < 2; ++ti)
    #pragma unroll
    for (int tj = 0; tj < 4; ++tj)
      #pragma unroll
      for (int r = 0; r < 8; ++r) {
        const int rr = row0 + wr * 32 + ti * 16 + r + (hiL ? 8 : 0);
        const int cc = col0 + wc * 64 + tj * 16 + ln;
        out[(size_t)rr * EMB + cc] = acc[ti][tj][r] + bo[cc];
      }
}

// ---------------------------------------------------------------------------
extern "C" void kernel_launch(void* const* d_in, const int* in_sizes, int n_in,
                              void* d_out, int out_size, void* d_ws, size_t ws_size,
                              hipStream_t stream)
{
  const float* x  = (const float*)d_in[0];
  const float* Wq = (const float*)d_in[1];
  const float* bq = (const float*)d_in[2];
  const float* Wk = (const float*)d_in[3];
  const float* bk = (const float*)d_in[4];
  const float* Wv = (const float*)d_in[5];
  const float* bv = (const float*)d_in[6];
  const float* Wo = (const float*)d_in[7];
  const float* bo = (const float*)d_in[8];

  char* ws = (char*)d_ws;
  const size_t SZ = (size_t)BQ * NH * SEQ * HD * 2;   // 16 MB per bf16 tensor
  __bf16* qbuf = (__bf16*)(ws);
  __bf16* kbuf = (__bf16*)(ws + SZ);
  __bf16* vbuf = (__bf16*)(ws + 2 * SZ);
  __bf16* abuf = (__bf16*)(ws + 3 * SZ);

  dim3 g1(BQ * SEQ / 128, EMB / 128, 3);
  qkv_rope_wmma<<<g1, 256, 0, stream>>>(x, Wq, bq, Wk, bk, Wv, bv, qbuf, kbuf, vbuf);

  dim3 g2(SEQ / 128, BQ * NH);
  const size_t lds2 = (size_t)(2 * 128 + 8 * 16) * KV_STRIDE * sizeof(__bf16); // 104448
  flash_attn_wmma<<<g2, 256, lds2, stream>>>(qbuf, kbuf, vbuf, abuf);

  dim3 g3(BQ * SEQ / 128, EMB / 128);
  out_proj_wmma<<<g3, 256, 0, stream>>>(abuf, Wo, bo, (float*)d_out);
}
